// GATLayer_7851200217745
// MI455X (gfx1250) — matrix operand order repaired and checked by reading, not compiled, not run on hardware
//
#include <hip/hip_runtime.h>

typedef __attribute__((ext_vector_type(16))) __bf16 v16bf;
typedef __attribute__((ext_vector_type(8)))  __bf16 bf16x8;
typedef __attribute__((ext_vector_type(2)))  __bf16 bf16x2;
typedef __attribute__((ext_vector_type(8)))  float  v8f;

#define KN     16
#define KP1    17
#define FDIM   128
#define NHEAD  8
#define DHEAD  64
#define TM     16          // nodes per workgroup (one WMMA M-tile)
#define FP     136         // padded bf16 row length (128 + 8)
#define NEG_SLOPE 0.2f
#define WSB_ELEMS (NHEAD * FDIM * DHEAD)   // 65536 bf16 = 128 KB

// ---------------------------------------------------------------------------
// Pre-pass: convert W [H][F][D] fp32 -> bf16 in WMMA-B lane layout in d_ws.
// wsB[(((h*4+dt)*4+kk)*32 + lane)*16 + i]:
//   N(col d) = dt*16 + (lane&15),  K(f) = kk*32 + (lane>>4)*16 + i
// ---------------------------------------------------------------------------
__global__ __launch_bounds__(256)
void prep_w_kernel(const float* __restrict__ W, __bf16* __restrict__ wsB)
{
    int e = blockIdx.x * 256 + threadIdx.x;      // 0 .. 65535
    int i    = e & 15;
    int lane = (e >> 4) & 31;
    int kk   = (e >> 9) & 3;
    int dt   = (e >> 11) & 3;
    int h    = e >> 13;
    int K = kk * 32 + (lane >> 4) * 16 + i;
    int d = dt * 16 + (lane & 15);
    wsB[e] = (__bf16)W[((size_t)h * FDIM + K) * DHEAD + d];
}

__global__ __launch_bounds__(256)
void gat_fused_kernel(const float* __restrict__ x_self,
                      const float* __restrict__ x_neigh,
                      const float* __restrict__ W,
                      const float* __restrict__ a_self,
                      const float* __restrict__ a_neigh,
                      const __bf16* __restrict__ wsB, int useWs,
                      float* __restrict__ out)
{
    // ~128 KB LDS -> 2 workgroups / WGP (320 KB)
    __shared__ __bf16 sX[KP1][TM][FP];         // x_all tile, [k][n][f]  (~72.3 KB)
    __shared__ __bf16 sABn[4][32][16];         // a_neigh B-tiles (4 KB)
    __shared__ __bf16 sASl[4][32][16];         // a_self  B-tiles (4 KB)
    __shared__ float  sLog[18][16][16];        // logits [j][h][n]; j=17 is p_self (18 KB)
    __shared__ __bf16 sXa[NHEAD][TM][FP];      // aggregated feats (~34 KB)

    const int tid  = threadIdx.x;
    const int lane = tid & 31;
    const int wave = tid >> 5;
    const int base = blockIdx.x * TM;
    const int rowM = lane & 15;
    const int grp  = lane >> 4;
    const int col  = lane & 15;

    // ---------------- build a_self / a_neigh WMMA-B tiles (heads 8..15 = 0) ----------------
    for (int e = tid; e < 4 * 32 * 16; e += 256) {
        int kk = e >> 9;
        int L  = (e >> 4) & 31;
        int i  = e & 15;
        int hN = L & 15;
        int K  = kk * 32 + (L >> 4) * 16 + i;
        __bf16 vn = (__bf16)0.f, vs = (__bf16)0.f;
        if (hN < NHEAD) {
            vn = (__bf16)a_neigh[hN * FDIM + K];
            vs = (__bf16)a_self [hN * FDIM + K];
        }
        sABn[kk][L][i] = vn;
        sASl[kk][L][i] = vs;
    }

    // ---------------- stage x_self tile (k = 0 slot) ----------------
    for (int q = tid; q < TM * (FDIM / 4); q += 256) {
        int n  = q / (FDIM / 4);
        int fq = q % (FDIM / 4);
        float4 v = ((const float4*)(x_self + (size_t)(base + n) * FDIM))[fq];
        bf16x2 lo; lo.x = (__bf16)v.x; lo.y = (__bf16)v.y;
        bf16x2 hi; hi.x = (__bf16)v.z; hi.y = (__bf16)v.w;
        *(bf16x2*)&sX[0][n][fq * 4 + 0] = lo;
        *(bf16x2*)&sX[0][n][fq * 4 + 2] = hi;
    }

    // ---------------- stage x_neigh tile (k = 1..16), single HBM pass ----------------
    for (int q = tid; q < TM * KN * (FDIM / 4); q += 256) {
        int n   = q >> 9;                      // / 512
        int rem = q & 511;
        int k   = rem >> 5;                    // / 32
        int fq  = rem & 31;
        float4 v = ((const float4*)(x_neigh + ((size_t)(base + n) * KN + k) * FDIM))[fq];
        bf16x2 lo; lo.x = (__bf16)v.x; lo.y = (__bf16)v.y;
        bf16x2 hi; hi.x = (__bf16)v.z; hi.y = (__bf16)v.w;
        *(bf16x2*)&sX[k + 1][n][fq * 4 + 0] = lo;
        *(bf16x2*)&sX[k + 1][n][fq * 4 + 2] = hi;
    }
    __syncthreads();

    // ---------------- attention logits via WMMA ----------------
    // Rows r = k*16+n, so M-tile j == k-slice sX[j].  Jobs 0..16: x_all·a_neigh^T.
    // Job 17: x_self·a_self^T.  D stored transposed -> sLog[j][col][m], branch-free.
    for (int j = wave; j < 18; j += 8) {
        int ksl = (j < 17) ? j : 0;
        const __bf16* aTile = (j < 17) ? &sABn[0][0][0] : &sASl[0][0][0];
        v8f acc = {};
        #pragma unroll
        for (int kk = 0; kk < 4; ++kk) {
            bf16x8 a0 = *(const bf16x8*)&sX[ksl][rowM][kk * 32 + grp * 16];
            bf16x8 a1 = *(const bf16x8*)&sX[ksl][rowM][kk * 32 + grp * 16 + 8];
            v16bf A = __builtin_shufflevector(a0, a1, 0,1,2,3,4,5,6,7,8,9,10,11,12,13,14,15);
            const __bf16* bt = aTile + ((kk * 32 + lane) << 4);
            bf16x8 b0 = *(const bf16x8*)bt;
            bf16x8 b1 = *(const bf16x8*)(bt + 8);
            v16bf B = __builtin_shufflevector(b0, b1, 0,1,2,3,4,5,6,7,8,9,10,11,12,13,14,15);
            acc = __builtin_amdgcn_wmma_f32_16x16x32_bf16(
                false, A, false, B, (short)0, acc, false, false);
        }
        float4 lo = { acc[0], acc[1], acc[2], acc[3] };
        float4 hi = { acc[4], acc[5], acc[6], acc[7] };
        *(float4*)&sLog[j][col][grp * 8 + 0] = lo;
        *(float4*)&sLog[j][col][grp * 8 + 4] = hi;
    }
    __syncthreads();

    // ---------------- LeakyReLU + softmax over 17 neighbors, one thread per (n,h) ----------------
    if (tid < TM * NHEAD) {
        int n = tid >> 3, h = tid & 7;
        float ps = sLog[17][h][n];
        float l[KP1], m = -3.402823e38f;
        #pragma unroll
        for (int k = 0; k < KP1; ++k) {
            float v = sLog[k][h][n] + ps;
            v = (v >= 0.f) ? v : NEG_SLOPE * v;
            l[k] = v;
            m = (v > m) ? v : m;
        }
        float s = 0.f;
        #pragma unroll
        for (int k = 0; k < KP1; ++k) { l[k] = __expf(l[k] - m); s += l[k]; }
        float inv = 1.f / s;
        #pragma unroll
        for (int k = 0; k < KP1; ++k) sLog[k][h][n] = l[k] * inv;
    }
    __syncthreads();

    // ---------------- weighted aggregation: xa[h][n][f] = sum_k attn * x_all ----------------
    // Two threads per (n,h) pair; attention weights in registers; b128 LDS traffic.
    {
        int pair  = tid >> 1;                 // 0..127
        int n     = pair >> 3;
        int h     = pair & 7;
        int fbase = (tid & 1) * 64;
        float at[KP1];
        #pragma unroll
        for (int k = 0; k < KP1; ++k) at[k] = sLog[k][h][n];
        #pragma unroll 1
        for (int c = 0; c < 8; ++c) {
            int f0 = fbase + c * 8;
            float acc[8];
            #pragma unroll
            for (int j = 0; j < 8; ++j) acc[j] = 0.f;
            #pragma unroll
            for (int k = 0; k < KP1; ++k) {
                bf16x8 xv = *(const bf16x8*)&sX[k][n][f0];
                #pragma unroll
                for (int j = 0; j < 8; ++j) acc[j] += at[k] * (float)xv[j];
            }
            bf16x8 o;
            #pragma unroll
            for (int j = 0; j < 8; ++j) o[j] = (__bf16)acc[j];
            *(bf16x8*)&sXa[h][n][f0] = o;
        }
    }
    __syncthreads();

    // ---------------- projection GEMM via WMMA: out[n, h*64+d] = xa[h][n][:] @ W[h] ----------------
    // Wave w owns head h=w, d-tiles 0..3, 4 K-steps of 32 each.
    #pragma unroll
    for (int i = 0; i < 4; ++i) {
        int h  = wave;                        // wave-uniform
        int dt = i;
        const float* Wh = W + (size_t)h * FDIM * DHEAD + dt * 16 + col;

        v8f acc = {};
        #pragma unroll
        for (int kk = 0; kk < 4; ++kk) {
            bf16x8 a0 = *(const bf16x8*)&sXa[h][rowM][kk * 32 + grp * 16];
            bf16x8 a1 = *(const bf16x8*)&sXa[h][rowM][kk * 32 + grp * 16 + 8];
            v16bf A = __builtin_shufflevector(a0, a1, 0,1,2,3,4,5,6,7,8,9,10,11,12,13,14,15);
            v16bf B;
            if (useWs) {
                const __bf16* bt = wsB + (size_t)((((h * 4 + dt) * 4 + kk) * 32 + lane) << 4);
                bf16x8 b0 = *(const bf16x8*)bt;
                bf16x8 b1 = *(const bf16x8*)(bt + 8);
                B = __builtin_shufflevector(b0, b1, 0,1,2,3,4,5,6,7,8,9,10,11,12,13,14,15);
            } else {
                v16bf Bt = {};
                #pragma unroll
                for (int v = 0; v < 8; ++v) {
                    int kB = kk * 32 + grp * 16 + 2 * v;
                    Bt[2 * v + 0] = (__bf16)Wh[(size_t)(kB + 0) * DHEAD];
                    Bt[2 * v + 1] = (__bf16)Wh[(size_t)(kB + 1) * DHEAD];
                }
                B = Bt;
            }
            acc = __builtin_amdgcn_wmma_f32_16x16x32_bf16(
                false, A, false, B, (short)0, acc, false, false);
        }
        #pragma unroll
        for (int r = 0; r < 8; ++r) {
            int m = r + (grp << 3);
            out[(size_t)(base + m) * (NHEAD * DHEAD) + h * DHEAD + dt * 16 + col] = acc[r];
        }
    }
}

extern "C" void kernel_launch(void* const* d_in, const int* in_sizes, int n_in,
                              void* d_out, int out_size, void* d_ws, size_t ws_size,
                              hipStream_t stream) {
    const float* x_self  = (const float*)d_in[0];
    const float* x_neigh = (const float*)d_in[1];
    const float* W       = (const float*)d_in[2];
    const float* a_self  = (const float*)d_in[3];
    const float* a_neigh = (const float*)d_in[4];
    float* out = (float*)d_out;

    int useWs = (ws_size >= (size_t)WSB_ELEMS * sizeof(__bf16)) ? 1 : 0;
    __bf16* wsB = (__bf16*)d_ws;
    if (useWs) {
        prep_w_kernel<<<WSB_ELEMS / 256, 256, 0, stream>>>(W, wsB);
    }

    int n = in_sizes[0] / FDIM;              // 50000
    int grid = (n + TM - 1) / TM;            // 3125
    gat_fused_kernel<<<grid, 256, 0, stream>>>(x_self, x_neigh, W, a_self, a_neigh,
                                               wsB, useWs, out);
}
